// Image2SphereProjector_69037304315949
// MI455X (gfx1250) — matrix operand order repaired and checked by reading, not compiled, not run on hardware
//
#include <hip/hip_runtime.h>

typedef __attribute__((ext_vector_type(16))) __bf16 v16bf;
typedef __attribute__((ext_vector_type(8)))  float  v8f;
typedef __attribute__((ext_vector_type(4)))  float  v4f;

#define BB   32
#define CC   512
#define HH   64
#define WW   64
#define PP   96          // N of GEMM-1
#define NHC  49          // spherical-harmonic coeffs
#define K1   (HH*WW)     // 4096
#define M1   (BB*CC)     // 16384
#define KSTEPS (K1/32)   // 128
#define NT   (PP/16)     // 6 N-tiles of 16

// ---------------------------------------------------------------------------
// Prep: weight (H*W, P) f32 -> bf16, pre-swizzled into WMMA B-operand layout.
// ws dword index = ((kk*NT + t)*32 + lane)*8 + v
// element: K = kk*32 + (lane&16 ? 8 : 0) + 2v + (v>=4 ? 8 : 0) + h ,  N = t*16 + (lane&15)
// ---------------------------------------------------------------------------
__global__ __launch_bounds__(256) void prep_weight_bf16(
    const float* __restrict__ w, unsigned int* __restrict__ wsb) {
  int tid = blockIdx.x * 256 + threadIdx.x;          // 196608 threads total
  int v  = tid & 7;
  int l  = (tid >> 3) & 31;
  int t  = (tid >> 8) % NT;
  int kk = tid / (NT * 256);
  int k0 = kk * 32 + ((l & 16) ? 8 : 0) + 2 * v + ((v >= 4) ? 8 : 0);
  int n  = t * 16 + (l & 15);
  unsigned short u0 = __builtin_bit_cast(unsigned short, (__bf16)w[(size_t)k0 * PP + n]);
  unsigned short u1 = __builtin_bit_cast(unsigned short, (__bf16)w[(size_t)(k0 + 1) * PP + n]);
  wsb[tid] = ((unsigned int)u1 << 16) | (unsigned int)u0;
}

// ---------------------------------------------------------------------------
// x is streamed exactly once: use non-temporal loads so the 256 MB stream
// does not evict the hot 768 KB weight / 19 KB Y working set from L2.
// (__builtin_nontemporal_load needs a clang ext-vector type, not HIP float4.)
// ---------------------------------------------------------------------------
__device__ __forceinline__ v4f ldg_nt4(const float* p) {
  return __builtin_nontemporal_load((const v4f*)p);
}

__device__ __forceinline__ v16bf make_a(const float* p0) {
  v4f f0 = ldg_nt4(p0);
  v4f f1 = ldg_nt4(p0 + 4);
  v4f f2 = ldg_nt4(p0 + 16);
  v4f f3 = ldg_nt4(p0 + 20);
  v16bf a;
  a[0]  = (__bf16)f0[0]; a[1]  = (__bf16)f0[1]; a[2]  = (__bf16)f0[2]; a[3]  = (__bf16)f0[3];
  a[4]  = (__bf16)f1[0]; a[5]  = (__bf16)f1[1]; a[6]  = (__bf16)f1[2]; a[7]  = (__bf16)f1[3];
  a[8]  = (__bf16)f2[0]; a[9]  = (__bf16)f2[1]; a[10] = (__bf16)f2[2]; a[11] = (__bf16)f2[3];
  a[12] = (__bf16)f3[0]; a[13] = (__bf16)f3[1]; a[14] = (__bf16)f3[2]; a[15] = (__bf16)f3[3];
  return a;
}

// ---------------------------------------------------------------------------
// Fused: z = relu(X @ Wbf16) via WMMA bf16, then out = (z @ Y)/sqrt(P) in f32.
// 2 waves / block; each wave owns a 32-row M tile (2 WMMA row-tiles) so every
// B-operand fetch feeds 2 WMMAs -> halves L2 weight traffic.
// ---------------------------------------------------------------------------
__global__ __launch_bounds__(64) void fused_proj(
    const float* __restrict__ x, const v16bf* __restrict__ wb,
    const float* __restrict__ Y, float* __restrict__ out) {
  __shared__ float sY[PP * NHC];          // 96*49 f32 = 18816 B, block-shared
  __shared__ float sz[2][32][PP + 1];     // per-wave relu(z) tile, stride 97

  const int lane = threadIdx.x & 31;
  const int wv   = threadIdx.x >> 5;
  const int tile = blockIdx.x * 2 + wv;   // 512 row-tiles of 32

  for (int i = threadIdx.x; i < PP * NHC; i += 64) sY[i] = Y[i];

  // A-operand addressing per CDNA5 16-bit A layout:
  // lane holds row M = lane%16, K runs [b,b+8) and [b+16,b+24), b = lane<16?0:8
  const float* xr0 = x + (size_t)(tile * 32 + (lane & 15)) * K1;
  const float* xr1 = xr0 + (size_t)16 * K1;
  const int kb = (lane & 16) ? 8 : 0;

  v8f acc0[NT] = {};
  v8f acc1[NT] = {};

  #pragma unroll 1
  for (int kk = 0; kk < KSTEPS; ++kk) {
    const int ko = kk * 32 + kb;
    v16bf a0 = make_a(xr0 + ko);
    v16bf a1 = make_a(xr1 + ko);

    const v16bf* bp = wb + (size_t)(kk * NT) * 32 + lane;  // 32B contiguous/lane
    #pragma unroll
    for (int t = 0; t < NT; ++t) {
      v16bf b = bp[t * 32];
      acc0[t] = __builtin_amdgcn_wmma_f32_16x16x32_bf16(
          false, a0, false, b, (short)0, acc0[t], false, false);
      acc1[t] = __builtin_amdgcn_wmma_f32_16x16x32_bf16(
          false, a1, false, b, (short)0, acc1[t], false, false);
    }
  }

  // Accumulator layout: lane holds column p = t*16 + lane%16,
  // rows M = r + (lane<16 ? 0 : 8) in vgpr r. Spill relu(z) to LDS.
  {
    const int p  = (lane & 15);
    const int mb = (lane & 16) ? 8 : 0;
    #pragma unroll
    for (int t = 0; t < NT; ++t)
      #pragma unroll
      for (int r = 0; r < 8; ++r) {
        sz[wv][mb + r][t * 16 + p]      = fmaxf(acc0[t][r], 0.0f);
        sz[wv][16 + mb + r][t * 16 + p] = fmaxf(acc1[t][r], 0.0f);
      }
  }
  __syncthreads();

  // Stage 2: out[m][i] = (1/sqrt(96)) * sum_p relu(z)[m][p] * Y[p][i], full f32
  const float scale = 0.10206207261596575f;  // 1/sqrt(96)
  for (int idx = lane; idx < 32 * NHC; idx += 32) {
    int m = idx / NHC;
    int i = idx - m * NHC;
    float s = 0.0f;
    #pragma unroll 4
    for (int p = 0; p < PP; ++p) s += sz[wv][m][p] * sY[p * NHC + i];
    __builtin_nontemporal_store(s * scale,
                                out + (size_t)(tile * 32 + m) * NHC + i);
  }
}

// ---------------------------------------------------------------------------
extern "C" void kernel_launch(void* const* d_in, const int* in_sizes, int n_in,
                              void* d_out, int out_size, void* d_ws, size_t ws_size,
                              hipStream_t stream) {
  const float* x = (const float*)d_in[0];   // (32,512,64,64)
  const float* w = (const float*)d_in[1];   // (64,64,96)
  const float* Y = (const float*)d_in[2];   // (96,49)
  float* out = (float*)d_out;               // (32,512,49)
  unsigned int* wsb = (unsigned int*)d_ws;  // needs 768 KB: bf16 swizzled weight

  // 128 ksteps * 6 tiles * 32 lanes * 8 dwords = 196608 dwords
  prep_weight_bf16<<<768, 256, 0, stream>>>(w, wsb);
  // 512 row-tiles of 32 rows / 2 waves per block = 256 blocks of 64 threads
  fused_proj<<<256, 64, 0, stream>>>(x, (const v16bf*)d_ws, Y, out);
}
